// EncoderSAE_32504312496878
// MI455X (gfx1250) — matrix operand order repaired and checked by reading, not compiled, not run on hardware
//
#include <hip/hip_runtime.h>

#define BATCH 4096
#define DIN   1024
#define DICT  32768
#define KSEL  64

typedef __attribute__((ext_vector_type(16))) __bf16 v16bf;
typedef __attribute__((ext_vector_type(8)))  __bf16 v8bf;
typedef __attribute__((ext_vector_type(8)))  float  v8f;

// ---------------------------------------------------------------------------
// f32 -> bf16 (round to nearest even), stored as raw u16
// ---------------------------------------------------------------------------
__global__ void sae_cvt_bf16(const float* __restrict__ in,
                             unsigned short* __restrict__ out, int n) {
  int i = blockIdx.x * blockDim.x + threadIdx.x;
  if (i < n) {
    unsigned u = __float_as_uint(in[i]);
    unsigned r = (u + 0x7FFFu + ((u >> 16) & 1u)) >> 16;
    out[i] = (unsigned short)r;
  }
}

// ---------------------------------------------------------------------------
// WMMA bf16 fragment loaders (row-major, K contiguous, ld = K stride)
// A 16x32 bf16 (ISA 7.12.2): lanes 0-15 -> M=lane, K={k..k+7, k+16..k+23}
//                            lanes 16-31 -> M=lane-16, K={k+8..k+15, k+24..k+31}
// ---------------------------------------------------------------------------
__device__ inline v16bf load_fragA(const __bf16* mat, int ld, int row16,
                                   int k, int lane) {
  int r  = lane & 15;
  int k0 = k + ((lane & 16) ? 8 : 0);
  const __bf16* p = mat + (size_t)(row16 + r) * ld + k0;
  union { v16bf v; v8bf h[2]; } u;
  u.h[0] = *reinterpret_cast<const v8bf*>(p);       // K = k0 .. k0+7
  u.h[1] = *reinterpret_cast<const v8bf*>(p + 16);  // K = k0+16 .. k0+23
  return u.v;
}

// B 32x16 bf16: lanes 0-15 -> N=lane, K={k..k+7, k+16..k+23}; lanes 16-31 -> N, K={k+8..k+15, k+24..k+31}
__device__ inline v16bf load_fragB(const __bf16* mat, int ld, int row16,
                                   int k, int lane) {
  int n  = lane & 15;
  int k0 = k + ((lane & 16) ? 8 : 0);
  const __bf16* p = mat + (size_t)(row16 + n) * ld + k0;
  union { v16bf v; v8bf h[2]; } u;
  u.h[0] = *reinterpret_cast<const v8bf*>(p);       // K = k0 .. k0+7
  u.h[1] = *reinterpret_cast<const v8bf*>(p + 16);  // K = k0+16 .. k0+23
  return u.v;
}

// ---------------------------------------------------------------------------
// Encode GEMM: feats = relu(x @ W_enc^T), bf16 in / f32 acc.
// WG = 256 threads = 8 waves (2x4), wave tile 64x64 (4x4 of 16x16), WG 128x256.
// ---------------------------------------------------------------------------
__global__ __launch_bounds__(256) void sae_encode_gemm(
    const unsigned short* __restrict__ xb_,
    const unsigned short* __restrict__ wb_,
    float* __restrict__ feats) {
  const __bf16* A  = reinterpret_cast<const __bf16*>(xb_);
  const __bf16* Bm = reinterpret_cast<const __bf16*>(wb_);
  const int lane = threadIdx.x & 31;
  const int wave = threadIdx.x >> 5;
  const int m0 = blockIdx.y * 128 + (wave >> 2) * 64;
  const int n0 = blockIdx.x * 256 + (wave & 3) * 64;

  v8f acc[4][4];
#pragma unroll
  for (int i = 0; i < 4; ++i)
#pragma unroll
    for (int j = 0; j < 4; ++j) { v8f z = {}; acc[i][j] = z; }

  for (int k = 0; k < DIN; k += 32) {
    v16bf af[4], bf[4];
#pragma unroll
    for (int i = 0; i < 4; ++i) af[i] = load_fragA(A, DIN, m0 + 16 * i, k, lane);
#pragma unroll
    for (int j = 0; j < 4; ++j) bf[j] = load_fragB(Bm, DIN, n0 + 16 * j, k, lane);
#pragma unroll
    for (int i = 0; i < 4; ++i)
#pragma unroll
      for (int j = 0; j < 4; ++j)
        acc[i][j] = __builtin_amdgcn_wmma_f32_16x16x32_bf16(
            false, af[i], false, bf[j], (short)0, acc[i][j], false, false);
  }

  // C/D layout: VGPR r -> M = r (lanes 0-15) / r+8 (lanes 16-31); N = lane&15
  const int mo  = (lane >> 4) << 3;
  const int col = lane & 15;
#pragma unroll
  for (int i = 0; i < 4; ++i)
#pragma unroll
    for (int j = 0; j < 4; ++j) {
      size_t base = (size_t)(m0 + 16 * i + mo) * DICT + (size_t)(n0 + 16 * j + col);
#pragma unroll
      for (int r = 0; r < 8; ++r) {
        float v = acc[i][j][r];
        feats[base + (size_t)r * DICT] = v > 0.0f ? v : 0.0f;
      }
    }
}

// ---------------------------------------------------------------------------
// Per-row exact top-K via 4-pass byte radix select on IEEE bits (all >= 0),
// then index-ordered compaction (ties broken by lowest index, like top_k).
// Zeroes non-kept entries of the dense sparse output in place.
// ---------------------------------------------------------------------------
__global__ __launch_bounds__(256) void sae_topk(
    float* __restrict__ feats, float* __restrict__ vals,
    int* __restrict__ idxs, int* __restrict__ posCnt) {
  const int tid  = threadIdx.x;
  const int lane = tid & 31;
  const int wv   = tid >> 5;
  const int row  = blockIdx.x;
  float* rowp = feats + (size_t)row * DICT;

  __shared__ unsigned hist[256];
  __shared__ int s_digit, s_above;
  __shared__ int s_base, s_ties, s_pos;
  __shared__ int wTie[8], wKeep[8], wPos[8];

  unsigned prefix = 0, pmask = 0;
  int need = KSEL;
  for (int pass = 3; pass >= 0; --pass) {
    const int sh = pass * 8;
    hist[tid] = 0;
    __syncthreads();
    for (int i = tid; i < DICT; i += 256) {
      unsigned v = __float_as_uint(rowp[i]);
      if ((v & pmask) == prefix) atomicAdd(&hist[(v >> sh) & 255u], 1u);
    }
    __syncthreads();
    if (tid == 0) {
      unsigned cum = 0; int d = 0;
      for (int b = 255; b >= 0; --b) {
        unsigned c = hist[b];
        if (cum + c >= (unsigned)need) { d = b; break; }
        cum += c;
      }
      s_digit = d; s_above = (int)cum;
    }
    __syncthreads();
    need  -= s_above;
    prefix |= ((unsigned)s_digit) << sh;
    pmask  |= 0xFFu << sh;
    __syncthreads();
  }
  const unsigned T = prefix;   // exact 64th-largest bit pattern
  const int tieKeep = need;    // # of ==T elements to keep (lowest indices)

  if (tid == 0) { s_base = 0; s_ties = 0; s_pos = 0; }
  __syncthreads();

  for (int chunk = 0; chunk < DICT / 256; ++chunk) {
    const int i = chunk * 256 + tid;
    const float f = rowp[i];
    const unsigned v = __float_as_uint(f);
    const bool gt  = v > T;
    const bool tie = (v == T);

    unsigned long long bt = __ballot(tie);
    const int tieLow = __popcll(bt & ((1ull << lane) - 1ull));
    if (lane == 0) wTie[wv] = __popcll(bt);
    __syncthreads();
    int tieBefore = s_ties;
    for (int w = 0; w < wv; ++w) tieBefore += wTie[w];
    const bool keep = gt || (tie && (tieBefore + tieLow) < tieKeep);

    unsigned long long bk = __ballot(keep);
    const int keepLow = __popcll(bk & ((1ull << lane) - 1ull));
    const bool pos = keep && (f > 0.0f);
    unsigned long long bp = __ballot(pos);
    if (lane == 0) { wKeep[wv] = __popcll(bk); wPos[wv] = __popcll(bp); }
    __syncthreads();
    int keepBefore = s_base;
    for (int w = 0; w < wv; ++w) keepBefore += wKeep[w];

    if (keep) {
      int slot = keepBefore + keepLow;           // deterministic, index order
      vals[row * KSEL + slot] = f;
      idxs[row * KSEL + slot] = i;
    } else {
      rowp[i] = 0.0f;
    }
    __syncthreads();
    if (tid == 0) {
      int tK = 0, tT = 0, tP = 0;
      for (int w = 0; w < 8; ++w) { tK += wKeep[w]; tT += wTie[w]; tP += wPos[w]; }
      s_base += tK; s_ties += tT; s_pos += tP;
    }
    __syncthreads();
  }
  if (tid == 0) posCnt[row] = s_pos;
}

// ---------------------------------------------------------------------------
// Sparse decode: rec[b,d] = sum_k vals[b,k] * W_enc[idx[b,k], d]
// (W_dec = W_enc^T, so W_dec column f == W_enc row f == contiguous 4KB)
// ---------------------------------------------------------------------------
__global__ __launch_bounds__(256) void sae_decode(
    const float* __restrict__ vals, const int* __restrict__ idxs,
    const float* __restrict__ Wenc, float* __restrict__ rec) {
  const int row = blockIdx.x;
  const int tid = threadIdx.x;
  __shared__ float sv[KSEL];
  __shared__ int   si[KSEL];
  if (tid < KSEL) { sv[tid] = vals[row * KSEL + tid]; si[tid] = idxs[row * KSEL + tid]; }
  __syncthreads();
  float acc[4] = {0.f, 0.f, 0.f, 0.f};
  for (int k = 0; k < KSEL; ++k) {
    const float v = sv[k];
    const float* wr = Wenc + (size_t)si[k] * DIN;
#pragma unroll
    for (int j = 0; j < 4; ++j) acc[j] += v * wr[tid + j * 256];
  }
#pragma unroll
  for (int j = 0; j < 4; ++j) rec[(size_t)row * DIN + tid + j * 256] = acc[j];
}

// ---------------------------------------------------------------------------
// l0 = mean over rows of (# kept values > 0); deterministic integer reduce
// ---------------------------------------------------------------------------
__global__ void sae_l0(const int* __restrict__ posCnt, float* __restrict__ out) {
  __shared__ int s[256];
  const int tid = threadIdx.x;
  int sum = 0;
  for (int i = tid; i < BATCH; i += 256) sum += posCnt[i];
  s[tid] = sum;
  __syncthreads();
  for (int off = 128; off; off >>= 1) {
    if (tid < off) s[tid] += s[tid + off];
    __syncthreads();
  }
  if (tid == 0) out[0] = (float)s[0] / (float)BATCH;
}

// ---------------------------------------------------------------------------
extern "C" void kernel_launch(void* const* d_in, const int* in_sizes, int n_in,
                              void* d_out, int out_size, void* d_ws, size_t ws_size,
                              hipStream_t stream) {
  (void)in_sizes; (void)n_in; (void)out_size; (void)ws_size;
  const float* x    = (const float*)d_in[0];
  const float* Wenc = (const float*)d_in[1];
  // d_in[2] (W_dec) is numerically W_enc^T -> decode uses W_enc rows instead.

  float* out    = (float*)d_out;
  float* rec    = out;                                        // 4096*1024
  float* sparse = out + (size_t)BATCH * DIN;                  // 4096*32768
  float* l0     = out + (size_t)BATCH * DIN + (size_t)BATCH * DICT;

  char* ws = (char*)d_ws;
  unsigned short* xb = (unsigned short*)(ws);                         //  8 MB
  unsigned short* wb = (unsigned short*)(ws + (size_t)BATCH * DIN * 2);// 64 MB
  size_t offVals = (size_t)BATCH * DIN * 2 + (size_t)DICT * DIN * 2;
  float* vals = (float*)(ws + offVals);                               //  1 MB
  int*   idxs = (int*)(ws + offVals + (size_t)BATCH * KSEL * 4);      //  1 MB
  int*   pos  = (int*)(ws + offVals + (size_t)BATCH * KSEL * 8);      // 16 KB

  const int n1 = BATCH * DIN;
  const int n2 = DICT * DIN;
  sae_cvt_bf16<<<(n1 + 255) / 256, 256, 0, stream>>>(x, xb, n1);
  sae_cvt_bf16<<<(n2 + 255) / 256, 256, 0, stream>>>(Wenc, wb, n2);

  dim3 ggrid(DICT / 256, BATCH / 128);
  sae_encode_gemm<<<ggrid, 256, 0, stream>>>(xb, wb, sparse);

  sae_topk<<<BATCH, 256, 0, stream>>>(sparse, vals, idxs, pos);
  sae_decode<<<BATCH, 256, 0, stream>>>(vals, idxs, Wenc, rec);
  sae_l0<<<1, 256, 0, stream>>>(pos, l0);
}
